// Layer_61469571940688
// MI455X (gfx1250) — matrix operand order repaired and checked by reading, not compiled, not run on hardware
//
#include <hip/hip_runtime.h>
#include <hip/hip_bf16.h>
#include <stdint.h>

// ---------- problem constants ----------
#define IN_DIM   2048
#define OUT_DIM  2048
#define BASIS    64
#define BATCH    4
#define SEQ      4096
#define ROWS_PER_BLK 32       // 2 M-subtiles of 16 -> each B fragment feeds 2 WMMAs
#define THREADS  256          // 8 wave32 waves
#define EPS      1e-5f

typedef __bf16 bf16_t;
typedef bf16_t v16bf __attribute__((ext_vector_type(16)));
typedef float  v8f   __attribute__((ext_vector_type(8)));
typedef unsigned int u32;
typedef u32    v4u   __attribute__((ext_vector_type(4)));

// 32-byte carrier for building a v16bf fragment from two 16B loads
union FragCast {
    struct { v4u lo, hi; } u;
    v16bf v;
};

// LDS x-tile: 32 rows x 2048 bf16, row stride 4096 B (128 KB total), XOR-swizzled
// at 16B-chunk granularity: physical chunk = chunk ^ (row & 15), so the 16 lanes
// of an A-fragment (one row per lane, same logical chunk) hit 16 distinct
// 16B-aligned bank groups -> conflict-free ds_load_b128.
__device__ __forceinline__ const char* lds_chunk_ptr(const char* base, int row, int chunk) {
    return base + row * 4096 + ((chunk ^ (row & 15)) << 4);
}
__device__ __forceinline__ char* lds_chunk_ptr(char* base, int row, int chunk) {
    return base + row * 4096 + ((chunk ^ (row & 15)) << 4);
}

// A fragment (16x32 bf16, M = row, half = lane/16):
// elements 0..7  <-> K = k0 + half*8 + e        (contiguous 16B)
// elements 8..15 <-> K = k0 + 16 + half*8 + e-8 (contiguous 16B)
__device__ __forceinline__ v16bf load_a_frag(const char* ldsx, int row, int half, int k0) {
    FragCast c;
    int ck = (k0 >> 3) + half;
    c.u.lo = *(const v4u*)lds_chunk_ptr(ldsx, row, ck);
    c.u.hi = *(const v4u*)lds_chunk_ptr(ldsx, row, ck + 2);
    return c.v;
}

// B fragment (32x16 bf16, N = lane%16, koff = (lane/16)*8):
// elements 0..7 <-> K = k0 + koff + e, elements 8..15 <-> K = k0 + 16 + koff + e-8
__device__ __forceinline__ v16bf load_b_frag(const bf16_t* wrow_k) {
    FragCast c;
    c.u.lo = *(const v4u*)(wrow_k);
    c.u.hi = *(const v4u*)(wrow_k + 16);
    return c.v;
}

// ---------- pre-pass: fp32 -> bf16 weight conversion into workspace ----------
__global__ __launch_bounds__(256) void convert_weights(
    const float* __restrict__ W, const float* __restrict__ Wr,
    bf16_t* __restrict__ Wb, bf16_t* __restrict__ Wrb, int n)
{
    int i = (blockIdx.x * 256 + threadIdx.x) * 4;
    if (i >= n) return;
    float4 a = *(const float4*)(W + i);
    float4 b = *(const float4*)(Wr + i);
    union { bf16_t h[4]; uint2 u2; } pa, pb;
    pa.h[0] = (bf16_t)a.x; pa.h[1] = (bf16_t)a.y; pa.h[2] = (bf16_t)a.z; pa.h[3] = (bf16_t)a.w;
    pb.h[0] = (bf16_t)b.x; pb.h[1] = (bf16_t)b.y; pb.h[2] = (bf16_t)b.z; pb.h[3] = (bf16_t)b.w;
    *(uint2*)(Wb  + i) = pa.u2;
    *(uint2*)(Wrb + i) = pb.u2;
}

// ---------- fused kernel: one block = 32 rows x full 2048 columns ----------
__global__ __launch_bounds__(THREADS) void fused_proj_ln_basis(
    const float*  __restrict__ x,
    const bf16_t* __restrict__ Wb,
    const bf16_t* __restrict__ Wrb,
    const float*  __restrict__ gamma,
    const float*  __restrict__ beta,
    const float*  __restrict__ Acoeff,   // (OUT_DIM, BASIS)
    const float*  __restrict__ Bbasis,   // (BASIS, OUT_DIM)
    float*        __restrict__ out)
{
    extern __shared__ char ldsx[];       // 32 * 4096 B = 128 KB swizzled bf16 x-tile
    __shared__ float s_sum[32], s_sq[32], s_sgb[32], s_gb[32], s_bb[32], s_scalar[32];

    const int tid  = threadIdx.x;
    const int wave = tid >> 5;
    const int lane = tid & 31;
    const int nloc = lane & 15;
    const int half = lane >> 4;
    const int m0   = blockIdx.x * ROWS_PER_BLK;   // global row base (row = b*SEQ + s)

    if (tid < 32) { s_sum[tid] = 0.f; s_sq[tid] = 0.f; s_sgb[tid] = 0.f; s_gb[tid] = 0.f; s_bb[tid] = 0.f; }

    // ---- stage x tile to LDS as swizzled bf16 ----
    for (int idx = tid * 4; idx < ROWS_PER_BLK * IN_DIM; idx += THREADS * 4) {
        int r = idx >> 11;                 // idx / 2048
        int c = idx & (IN_DIM - 1);
        float4 f = *(const float4*)(x + (size_t)(m0 + r) * IN_DIM + c);
        union { bf16_t h[4]; uint2 u2; } pk;
        pk.h[0] = (bf16_t)f.x; pk.h[1] = (bf16_t)f.y; pk.h[2] = (bf16_t)f.z; pk.h[3] = (bf16_t)f.w;
        char* dst = lds_chunk_ptr(ldsx, r, c >> 3) + (c & 7) * 2;  // 8B within 16B chunk
        *(uint2*)dst = pk.u2;
    }
    __syncthreads();

    // ---- per-row constants gb = sum γ·B[j], bb = sum β·B[j] (8 threads/row) ----
    {
        int r = tid >> 3;                  // 0..31
        int j = (m0 + r) & (BASIS - 1);    // SEQ % BASIS == 0, so s%64 == row%64
        const float* Bj = Bbasis + (size_t)j * OUT_DIM;
        float gb = 0.f, bb = 0.f;
        for (int n = tid & 7; n < OUT_DIM; n += 8) {
            float b = Bj[n];
            gb += gamma[n] * b;
            bb += beta[n]  * b;
        }
        atomicAdd(&s_gb[r], gb);
        atomicAdd(&s_bb[r], bb);
    }

    // ---- pass 1: transformed = x·Wᵀ, reduced into per-lane register partials ----
    // subtile 0 = rows 0..15, subtile 1 = rows 16..31; one B fragment feeds both.
    float ps0[8] = {}, pq0[8] = {}, pg0[8] = {};
    float ps1[8] = {}, pq1[8] = {}, pg1[8] = {};
    for (int t = 0; t < 16; ++t) {
        const int n0 = wave * 256 + t * 16;
        const int n  = n0 + nloc;

        // issue epilogue operands early so they resolve under the K-loop
        float g = gamma[n];
        float bv0[8], bv1[8];
        #pragma unroll
        for (int e = 0; e < 8; ++e) {
            int j0 = (m0 + e + half * 8) & (BASIS - 1);
            int j1 = (m0 + 16 + e + half * 8) & (BASIS - 1);
            bv0[e] = Bbasis[(size_t)j0 * OUT_DIM + n];
            bv1[e] = Bbasis[(size_t)j1 * OUT_DIM + n];
        }

        v8f acc0 = {}, acc1 = {};
        const bf16_t* wrow = Wb + (size_t)n * IN_DIM + half * 8;
        #pragma unroll 2
        for (int k0 = 0; k0 < IN_DIM; k0 += 32) {
            v16bf b  = load_b_frag(wrow + k0);
            v16bf a0 = load_a_frag(ldsx, nloc,      half, k0);
            v16bf a1 = load_a_frag(ldsx, nloc + 16, half, k0);
            acc0 = __builtin_amdgcn_wmma_f32_16x16x32_bf16(
                       false, a0, false, b, (short)0, acc0, false, false);
            acc1 = __builtin_amdgcn_wmma_f32_16x16x32_bf16(
                       false, a1, false, b, (short)0, acc1, false, false);
        }
        #pragma unroll
        for (int e = 0; e < 8; ++e) {
            float t0 = acc0[e], t1 = acc1[e];
            ps0[e] += t0;        ps1[e] += t1;
            pq0[e] += t0 * t0;   pq1[e] += t1 * t1;
            pg0[e] += t0 * g * bv0[e];
            pg1[e] += t1 * g * bv1[e];
        }
    }
    // one atomic flush per lane (48 ds_add_f32 total)
    #pragma unroll
    for (int e = 0; e < 8; ++e) {
        int M = e + half * 8;
        atomicAdd(&s_sum[M],      ps0[e]);
        atomicAdd(&s_sq[M],       pq0[e]);
        atomicAdd(&s_sgb[M],      pg0[e]);
        atomicAdd(&s_sum[M + 16], ps1[e]);
        atomicAdd(&s_sq[M + 16],  pq1[e]);
        atomicAdd(&s_sgb[M + 16], pg1[e]);
    }
    __syncthreads();

    // ---- finalize per-row scalar (LayerNorm folded analytically) ----
    if (tid < 32) {
        float mean = s_sum[tid] * (1.0f / OUT_DIM);
        float var  = s_sq[tid] * (1.0f / OUT_DIM) - mean * mean;
        float rs   = rsqrtf(var + EPS);
        s_scalar[tid] = rs * (s_sgb[tid] - mean * s_gb[tid]) + s_bb[tid];
    }
    __syncthreads();

    // ---- pass 2: residual = x·W_resᵀ fused with out = scalar·A[:,j] + residual ----
    for (int t = 0; t < 16; ++t) {
        const int n0 = wave * 256 + t * 16;
        const int n  = n0 + nloc;

        // hoist Acoeff gathers above the K-loop
        float av0[8], av1[8];
        #pragma unroll
        for (int e = 0; e < 8; ++e) {
            int j0 = (m0 + e + half * 8) & (BASIS - 1);
            int j1 = (m0 + 16 + e + half * 8) & (BASIS - 1);
            av0[e] = Acoeff[(size_t)n * BASIS + j0];
            av1[e] = Acoeff[(size_t)n * BASIS + j1];
        }

        v8f acc0 = {}, acc1 = {};
        const bf16_t* wrow = Wrb + (size_t)n * IN_DIM + half * 8;
        #pragma unroll 2
        for (int k0 = 0; k0 < IN_DIM; k0 += 32) {
            v16bf b  = load_b_frag(wrow + k0);
            v16bf a0 = load_a_frag(ldsx, nloc,      half, k0);
            v16bf a1 = load_a_frag(ldsx, nloc + 16, half, k0);
            acc0 = __builtin_amdgcn_wmma_f32_16x16x32_bf16(
                       false, a0, false, b, (short)0, acc0, false, false);
            acc1 = __builtin_amdgcn_wmma_f32_16x16x32_bf16(
                       false, a1, false, b, (short)0, acc1, false, false);
        }
        #pragma unroll
        for (int e = 0; e < 8; ++e) {
            int M = e + half * 8;
            out[(size_t)(m0 + M) * OUT_DIM + n]      = acc0[e] + s_scalar[M]      * av0[e];
            out[(size_t)(m0 + 16 + M) * OUT_DIM + n] = acc1[e] + s_scalar[M + 16] * av1[e];
        }
    }
}

extern "C" void kernel_launch(void* const* d_in, const int* in_sizes, int n_in,
                              void* d_out, int out_size, void* d_ws, size_t ws_size,
                              hipStream_t stream) {
    const float* x      = (const float*)d_in[0];
    const float* W      = (const float*)d_in[1];
    const float* W_res  = (const float*)d_in[2];
    const float* gamma  = (const float*)d_in[3];
    const float* beta   = (const float*)d_in[4];
    const float* Acoeff = (const float*)d_in[5];
    const float* Bbasis = (const float*)d_in[6];
    float* out = (float*)d_out;

    bf16_t* Wb  = (bf16_t*)d_ws;
    bf16_t* Wrb = Wb + (size_t)OUT_DIM * IN_DIM;

    const int nw = OUT_DIM * IN_DIM;                 // 4M elements
    convert_weights<<<nw / (256 * 4), 256, 0, stream>>>(W, W_res, Wb, Wrb, nw);

    const int nblocks = (BATCH * SEQ) / ROWS_PER_BLK;         // 512
    const size_t shmem = (size_t)ROWS_PER_BLK * 4096;         // 128 KB swizzled x tile
    // CDNA5 WGP has 320 KB LDS; opt in to >64 KB dynamic LDS (graph-capture safe,
    // not a stream operation, idempotent across calls).
    hipFuncSetAttribute((const void*)fused_proj_ln_basis,
                        hipFuncAttributeMaxDynamicSharedMemorySize, (int)shmem);
    fused_proj_ln_basis<<<nblocks, THREADS, shmem, stream>>>(
        x, Wb, Wrb, gamma, beta, Acoeff, Bbasis, out);
}
